// GeoGATLayer_13134009991334
// MI455X (gfx1250) — matrix operand order repaired and checked by reading, not compiled, not run on hardware
//
#include <hip/hip_runtime.h>
#include <hip/hip_bf16.h>

// GeoGAT fused attention for MI455X (gfx1250, wave32, WMMA bf16).
// B=8, N=2048, F=256.
//
// Pipeline:
//   P1: Gm = mask/(D'+eps), Msk = mask   (batch-invariant, computed once)
//   P2: X,W -> bf16                      (kills cvt in GEMM hot loop)
//   K1: hT[b][f][n] = bf16(X@W^T + b)    (WMMA bf16, f32 accum)
//   K2: s1,s2 row scores
//   K3: per-row softmax max/sum          (online, wave per row)
//   K4: out = (exp(e-max) @ h) / sum     (P built in regs, WMMA bf16)

#define B_ 8
#define N_ 2048
#define F_ 256

typedef __attribute__((ext_vector_type(16))) __bf16 v16bf;
typedef __attribute__((ext_vector_type(8)))  __bf16 v8bf;
typedef __attribute__((ext_vector_type(8)))  float  v8f;
typedef __attribute__((ext_vector_type(4)))  float  v4f;

union FragU { v16bf v; v8bf h[2]; };

// Load 8 consecutive f32 (32B, aligned) and round to 8 bf16.
__device__ inline v8bf cvt8(const float* __restrict__ p) {
  v4f x0 = ((const v4f*)p)[0];
  v4f x1 = ((const v4f*)p)[1];
  v8bf r;
  r[0] = (__bf16)x0.x; r[1] = (__bf16)x0.y; r[2] = (__bf16)x0.z; r[3] = (__bf16)x0.w;
  r[4] = (__bf16)x1.x; r[5] = (__bf16)x1.y; r[6] = (__bf16)x1.z; r[7] = (__bf16)x1.w;
  return r;
}

// ---------------------------------------------------------------------------
// P1: batch-invariant attention terms. Gm = mask * 1/(D'+eps), Msk = mask.
// 4 elements per thread, fully vectorized.
// ---------------------------------------------------------------------------
__global__ __launch_bounds__(256) void prep_mask_kernel(
    const float* __restrict__ Ag, const float* __restrict__ Dm,
    const float* __restrict__ thr_p, float* __restrict__ Gm,
    float* __restrict__ Msk) {
  const int idx = blockIdx.x * 256 + threadIdx.x;
  const size_t j4 = (size_t)idx * 4;
  const int i = (int)(j4 >> 11);          // row
  const int j = (int)(j4 & (N_ - 1));     // first col of the 4
  const float thr = thr_p[0];
  const v4f d = *(const v4f*)(Dm + j4);
  const v4f a = *(const v4f*)(Ag + j4);
  v4f g, k;
#pragma unroll
  for (int q = 0; q < 4; ++q) {
    float dd = d[q];
    if (j + q == i) dd = 1.0f;                               // diag := 1.0
    const float inv = 1.0f / (dd + 1e-5f);
    const float mk = 1.0f / (1.0f + __expf(-10.0f * (a[q] - thr)));
    g[q] = inv * mk;
    k[q] = mk;
  }
  *(v4f*)(Gm + j4) = g;
  *(v4f*)(Msk + j4) = k;
}

// P2: f32 -> bf16 bulk convert (8 elems/thread).
__global__ __launch_bounds__(256) void cvt_bf16_kernel(
    const float* __restrict__ src, __bf16* __restrict__ dst) {
  const size_t i8 = ((size_t)blockIdx.x * 256 + threadIdx.x) * 8;
  *(v8bf*)(dst + i8) = cvt8(src + i8);
}

// ---------------------------------------------------------------------------
// K1: h = X @ W^T + Wb, stored transposed bf16: hT[b][f][n].
// One wave => 16 rows x 64 cols. Frag layouts as loaded here:
//   A 16x32 bf16: lane m=l&15 row, K = 16kh..+15 contiguous
//   B 32x16 bf16: lane n=l&15 col, K = 16kh..+15 contiguous
//   C/D f32:      lane col = l&15, rows = 8*(l>>4)+r
// ---------------------------------------------------------------------------
template <bool PRE>
__global__ __launch_bounds__(256) void h_gemm_kernel(
    const void* __restrict__ Xsrc, const void* __restrict__ Wsrc,
    const float* __restrict__ Wb, __bf16* __restrict__ hT) {
  const int wave = (blockIdx.x * blockDim.x + threadIdx.x) >> 5;
  const int lane = threadIdx.x & 31;
  const int colquad = wave & 3;
  const int rt      = wave >> 2;
  const int b       = rt >> 7;
  const int row0    = (rt & 127) << 4;
  const int m  = lane & 15;
  const int kh = lane >> 4;
  const int colbase = colquad * 64;

  v8f acc[4] = {};
  const size_t xoff = ((size_t)b * N_ + row0 + m) * F_;

  for (int k0 = 0; k0 < F_; k0 += 32) {
    FragU af;
    if constexpr (PRE) {
      const __bf16* xrow = (const __bf16*)Xsrc + xoff;
      af.h[0] = *(const v8bf*)(xrow + k0 + 16 * kh);
      af.h[1] = *(const v8bf*)(xrow + k0 + 16 * kh + 8);
    } else {
      const float* xrow = (const float*)Xsrc + xoff;
      af.h[0] = cvt8(xrow + k0 + 16 * kh);
      af.h[1] = cvt8(xrow + k0 + 16 * kh + 8);
    }
#pragma unroll
    for (int t = 0; t < 4; ++t) {
      const int o = colbase + t * 16 + m;
      FragU bf;
      if constexpr (PRE) {
        const __bf16* wrow = (const __bf16*)Wsrc + (size_t)o * F_ + k0 + 16 * kh;
        bf.v = *(const v16bf*)wrow;                    // 32B contiguous
      } else {
        const float* wrow = (const float*)Wsrc + (size_t)o * F_ + k0 + 16 * kh;
        bf.h[0] = cvt8(wrow);
        bf.h[1] = cvt8(wrow + 8);
      }
      acc[t] = __builtin_amdgcn_wmma_f32_16x16x32_bf16(
          false, af.v, false, bf.v, (short)0, acc[t], false, false);
    }
  }

#pragma unroll
  for (int t = 0; t < 4; ++t) {
    const int col = colbase + t * 16 + m;
    const float bias = Wb[col];
    v8bf o;
#pragma unroll
    for (int r = 0; r < 8; ++r) o[r] = (__bf16)(acc[t][r] + bias);
    __bf16* dst = hT + ((size_t)b * F_ + col) * N_ + row0 + 8 * kh;
    *(v8bf*)dst = o;
  }
}

// ---------------------------------------------------------------------------
// K2: s1[b,n] = h[b,n,:].a1 ; s2[b,n] = h[b,n,:].a2 (from hT, coalesced in n)
// ---------------------------------------------------------------------------
__global__ __launch_bounds__(256) void s_kernel(
    const __bf16* __restrict__ hT, const float* __restrict__ a1,
    const float* __restrict__ a2, float* __restrict__ s1,
    float* __restrict__ s2) {
  const int idx = blockIdx.x * 256 + threadIdx.x;   // b*N + n
  const int b = idx >> 11;
  const int n = idx & (N_ - 1);
  const __bf16* p = hT + (size_t)b * F_ * N_ + n;
  float acc1 = 0.f, acc2 = 0.f;
#pragma unroll 4
  for (int c = 0; c < F_; ++c) {
    const float hv = (float)p[(size_t)c * N_];
    acc1 += hv * a1[c];
    acc2 += hv * a2[c];
  }
  s1[idx] = acc1;
  s2[idx] = acc2;
}

__device__ inline float score_raw(float s1v, float s2v, float ab, float dd,
                                  float ag, float thr) {
  float t = s1v + s2v + ab;
  float e = t > 0.f ? t : 0.1f * t;
  e += 1.0f / (dd + 1e-5f);
  e *= 1.0f / (1.0f + __expf(-10.0f * (ag - thr)));
  return e;
}

__device__ inline float score_pre(float s1v, float s2v, float ab, float gm,
                                  float mk) {
  const float t = s1v + s2v + ab;
  const float l = t > 0.f ? t : 0.1f * t;
  return fmaf(l, mk, gm);
}

// ---------------------------------------------------------------------------
// K3: per-row online softmax stats. One wave per (b,i).
// ---------------------------------------------------------------------------
template <bool PRE>
__global__ __launch_bounds__(256) void stats_kernel(
    const float* __restrict__ P0,   // PRE ? Gm  : A_geo
    const float* __restrict__ P1,   // PRE ? Msk : Dm
    const float* __restrict__ s1, const float* __restrict__ s2,
    const float* __restrict__ attn_b, const float* __restrict__ thr_p,
    float* __restrict__ rowmax, float* __restrict__ rowsum) {
  const int wave = (blockIdx.x * blockDim.x + threadIdx.x) >> 5;  // b*N + i
  const int lane = threadIdx.x & 31;
  const int b = wave >> 11;
  const int i = wave & (N_ - 1);
  const float s1v = s1[wave];
  const float ab = attn_b[0];
  const float thr = PRE ? 0.f : thr_p[0];
  const float* r0 = P0 + (size_t)i * N_;
  const float* r1 = P1 + (size_t)i * N_;
  const float* s2b = s2 + b * N_;

  float mx = -3.4e38f, sum = 0.f;
  for (int j = lane; j < N_; j += 32) {
    float e;
    if constexpr (PRE) {
      e = score_pre(s1v, s2b[j], ab, r0[j], r1[j]);
    } else {
      const float dd = (j == i) ? 1.0f : r1[j];
      e = score_raw(s1v, s2b[j], ab, dd, r0[j], thr);
    }
    const float nm = fmaxf(mx, e);
    sum = sum * __expf(mx - nm) + __expf(e - nm);
    mx = nm;
  }
#pragma unroll
  for (int off = 16; off > 0; off >>= 1) {
    const float mo = __shfl_xor(mx, off, 32);
    const float so = __shfl_xor(sum, off, 32);
    const float nm = fmaxf(mx, mo);
    sum = sum * __expf(mx - nm) + so * __expf(mo - nm);
    mx = nm;
  }
  if (lane == 0) {
    rowmax[wave] = mx;
    rowsum[wave] = sum;
  }
}

// ---------------------------------------------------------------------------
// K4: out = softmax(e) @ h. P 16x32 bf16 A-frag built in regs, 4 WMMAs/step.
// ---------------------------------------------------------------------------
template <bool PRE>
__global__ __launch_bounds__(256) void av_kernel(
    const float* __restrict__ P0,   // PRE ? Gm  : A_geo
    const float* __restrict__ P1,   // PRE ? Msk : Dm
    const float* __restrict__ s1, const float* __restrict__ s2,
    const float* __restrict__ attn_b, const float* __restrict__ thr_p,
    const __bf16* __restrict__ hT, const float* __restrict__ rowmax,
    const float* __restrict__ rowsum, float* __restrict__ out) {
  const int wave = (blockIdx.x * 256 + threadIdx.x) >> 5;
  const int lane = threadIdx.x & 31;
  const int colquad = wave & 3;
  const int rt      = wave >> 2;
  const int b       = rt >> 7;
  const int row0    = (rt & 127) << 4;
  const int m  = lane & 15;
  const int kh = lane >> 4;
  const int i  = row0 + m;
  const int colbase = colquad * 64;

  const float ab = attn_b[0];
  const float thr = PRE ? 0.f : thr_p[0];
  const float s1v = s1[b * N_ + i];
  const float mx  = rowmax[b * N_ + i];
  const float* r0 = P0 + (size_t)i * N_;
  const float* r1 = P1 + (size_t)i * N_;
  const float* s2b = s2 + b * N_;

  v8f acc[4] = {};
  for (int j0 = 0; j0 < N_; j0 += 32) {
    if (j0 + 32 < N_) {                      // global_prefetch_b8 next rows
      __builtin_prefetch(r0 + j0 + 32 + (lane & 15) * 2, 0, 3);
      __builtin_prefetch(r1 + j0 + 32 + (lane & 15) * 2, 0, 3);
    }
    FragU pf;
#pragma unroll
    for (int half = 0; half < 2; ++half) {
      const int jb = j0 + 16 * kh + 8 * half;  // A-frag K block for lane half
      const v4f x0 = *(const v4f*)(r0 + jb);
      const v4f x1 = *(const v4f*)(r0 + jb + 4);
      const v4f y0 = *(const v4f*)(r1 + jb);
      const v4f y1 = *(const v4f*)(r1 + jb + 4);
      const v4f z0 = *(const v4f*)(s2b + jb);
      const v4f z1 = *(const v4f*)(s2b + jb + 4);
      v8bf pk;
#pragma unroll
      for (int q = 0; q < 8; ++q) {
        const float u0 = (q < 4) ? x0[q] : x1[q - 4];
        const float u1 = (q < 4) ? y0[q] : y1[q - 4];
        const float s2v = (q < 4) ? z0[q] : z1[q - 4];
        float e;
        if constexpr (PRE) {
          e = score_pre(s1v, s2v, ab, u0, u1);
        } else {
          float dd = u1;
          if (jb + q == i) dd = 1.0f;
          e = score_raw(s1v, s2v, ab, dd, u0, thr);
        }
        pk[q] = (__bf16)__expf(e - mx);
      }
      pf.h[half] = pk;
    }
#pragma unroll
    for (int t = 0; t < 4; ++t) {
      const int c = colbase + t * 16 + m;
      const __bf16* hp = hT + ((size_t)b * F_ + c) * N_ + j0 + 16 * kh;
      const v16bf bfrag = *(const v16bf*)hp;
      acc[t] = __builtin_amdgcn_wmma_f32_16x16x32_bf16(
          false, pf.v, false, bfrag, (short)0, acc[t], false, false);
    }
  }

  float inv[8];
#pragma unroll
  for (int r = 0; r < 8; ++r)
    inv[r] = 1.0f / rowsum[b * N_ + row0 + 8 * kh + r];
#pragma unroll
  for (int t = 0; t < 4; ++t) {
    const int c = colbase + t * 16 + m;
#pragma unroll
    for (int r = 0; r < 8; ++r) {
      out[((size_t)b * N_ + row0 + 8 * kh + r) * F_ + c] = acc[t][r] * inv[r];
    }
  }
}

// ---------------------------------------------------------------------------
extern "C" void kernel_launch(void* const* d_in, const int* in_sizes, int n_in,
                              void* d_out, int out_size, void* d_ws,
                              size_t ws_size, hipStream_t stream) {
  const float* X    = (const float*)d_in[0];
  const float* Ag   = (const float*)d_in[1];
  const float* Dm   = (const float*)d_in[2];
  const float* Ww   = (const float*)d_in[3];
  const float* Wb   = (const float*)d_in[4];
  const float* a1   = (const float*)d_in[5];
  const float* a2   = (const float*)d_in[6];
  const float* atb  = (const float*)d_in[7];
  const float* thr  = (const float*)d_in[8];
  float* out = (float*)d_out;

  char* ws = (char*)d_ws;
  const size_t szXbf = (size_t)B_ * N_ * F_ * 2;   // 8 MB
  const size_t szWbf = (size_t)F_ * F_ * 2;        // 128 KB
  const size_t szhT  = (size_t)B_ * N_ * F_ * 2;   // 8 MB
  const size_t szNN  = (size_t)N_ * N_ * 4;        // 16 MB
  const size_t szS   = (size_t)B_ * N_ * 4;        // 64 KB
  const size_t full  = szXbf + szWbf + szhT + 2 * szNN + 4 * szS;

  if (ws_size >= full) {
    __bf16* Xbf = (__bf16*)ws;
    __bf16* Wbf = (__bf16*)(ws + szXbf);
    __bf16* hT  = (__bf16*)(ws + szXbf + szWbf);
    float*  Gm  = (float*)(ws + szXbf + szWbf + szhT);
    float*  Msk = Gm + (size_t)N_ * N_;
    float*  s1  = Msk + (size_t)N_ * N_;
    float*  s2  = s1 + B_ * N_;
    float*  rmax = s2 + B_ * N_;
    float*  rsum = rmax + B_ * N_;

    // P1: batch-invariant mask terms (N*N/4/256 = 4096 blocks).
    prep_mask_kernel<<<(N_ * (size_t)N_) / (4 * 256), 256, 0, stream>>>(
        Ag, Dm, thr, Gm, Msk);
    // P2: bf16 conversions.
    cvt_bf16_kernel<<<(B_ * N_ * F_) / (8 * 256), 256, 0, stream>>>(X, Xbf);
    cvt_bf16_kernel<<<(F_ * F_) / (8 * 256), 256, 0, stream>>>(Ww, Wbf);
    // K1..K4
    h_gemm_kernel<true><<<512, 256, 0, stream>>>(Xbf, Wbf, Wb, hT);
    s_kernel<<<(B_ * N_) / 256, 256, 0, stream>>>(hT, a1, a2, s1, s2);
    stats_kernel<true><<<(B_ * N_) / 8, 256, 0, stream>>>(Gm, Msk, s1, s2,
                                                          atb, thr, rmax, rsum);
    av_kernel<true><<<512, 256, 0, stream>>>(Gm, Msk, s1, s2, atb, thr, hT,
                                             rmax, rsum, out);
  } else {
    // Fallback: compute masks / conversions on the fly (small ws layout).
    __bf16* hT  = (__bf16*)ws;
    float*  s1  = (float*)(ws + szhT);
    float*  s2  = s1 + B_ * N_;
    float*  rmax = s2 + B_ * N_;
    float*  rsum = rmax + B_ * N_;

    h_gemm_kernel<false><<<512, 256, 0, stream>>>(X, Ww, Wb, hT);
    s_kernel<<<(B_ * N_) / 256, 256, 0, stream>>>(hT, a1, a2, s1, s2);
    stats_kernel<false><<<(B_ * N_) / 8, 256, 0, stream>>>(Ag, Dm, s1, s2,
                                                           atb, thr, rmax, rsum);
    av_kernel<false><<<512, 256, 0, stream>>>(Ag, Dm, s1, s2, atb, thr, hT,
                                              rmax, rsum, out);
  }
}
